// NumericalReasoning_40776419508954
// MI455X (gfx1250) — matrix operand order repaired and checked by reading, not compiled, not run on hardware
//
#include <hip/hip_runtime.h>
#include <math.h>

#define B_    16
#define S_    2048
#define H_    1024
#define N_    128
#define NREL_ 8

typedef __attribute__((ext_vector_type(16))) _Float16 v16h;
typedef __attribute__((ext_vector_type(8)))  _Float16 v8h;
typedef __attribute__((ext_vector_type(4)))  _Float16 v4h;
typedef __attribute__((ext_vector_type(8)))  float    v8f;

__device__ __forceinline__ v16h cat8(v8h lo, v8h hi) {
  return __builtin_shufflevector(lo, hi, 0, 1, 2, 3, 4, 5, 6, 7,
                                         8, 9, 10, 11, 12, 13, 14, 15);
}

__device__ __forceinline__ v8f wmma16(v16h a, v16h b, v8f c) {
  // D(16x16,f32) = A(16x32,f16) * B(32x16,f16) + C  -> v_wmma_f32_16x16x32_f16
  return __builtin_amdgcn_wmma_f32_16x16x32_f16(false, a, false, b, (short)0, c,
                                                false, false);
}

// ---------------------------------------------------------------------------
// Kernel 1: gather Init rows (store f16) + alpha = sigmoid(Init.w_alpha) in f32
// one 256-thread block per (b,n) row
// ---------------------------------------------------------------------------
__global__ void gather_alpha_kernel(const float* __restrict__ word_emb,
                                    const int*   __restrict__ num_ids,
                                    const float* __restrict__ w_alpha,
                                    const float* __restrict__ b_alpha,
                                    _Float16* __restrict__ Init16,
                                    float* __restrict__ alpha) {
  int bn  = blockIdx.x;            // 0 .. B*N-1
  int b   = bn / N_;
  int n   = bn % N_;
  int tid = threadIdx.x;           // 256 threads, 4 floats each

  int id  = num_ids[b * N_ + n];
  int idx = min(max(id, 0), S_ - 1);
  const float* src = word_emb + ((size_t)b * S_ + idx) * H_;

  float4 v = *(const float4*)(src + tid * 4);
  v4h hv = {(_Float16)v.x, (_Float16)v.y, (_Float16)v.z, (_Float16)v.w};
  *(v4h*)(Init16 + (size_t)bn * H_ + tid * 4) = hv;

  float4 w = *(const float4*)(w_alpha + tid * 4);
  float part = v.x * w.x + v.y * w.y + v.z * w.z + v.w * w.w;

  __shared__ float red[256];
  red[tid] = part;
  __syncthreads();
  for (int off = 128; off > 0; off >>= 1) {
    if (tid < off) red[tid] += red[tid + off];
    __syncthreads();
  }
  if (tid == 0) {
    float x = red[0] + b_alpha[0];
    alpha[b * N_ + n] = 1.0f / (1.0f + __expf(-x));
  }
}

// ---------------------------------------------------------------------------
// Kernel 2: A16[b,r,i,j] = (r==R(b,i,j)) * alpha_j * mask * (i!=j) / 127  (f16)
// ---------------------------------------------------------------------------
__global__ void build_A_kernel(const int*   __restrict__ num_ids,
                               const int*   __restrict__ is_response,
                               const float* __restrict__ numbers,
                               const float* __restrict__ alpha,
                               _Float16* __restrict__ A16) {
  int g   = blockIdx.x * blockDim.x + threadIdx.x;   // B*N*N
  int b   = g / (N_ * N_);
  int rem = g - b * (N_ * N_);
  int i   = rem / N_;
  int j   = rem % N_;

  int gt = numbers[b * N_ + i] > numbers[b * N_ + j];
  int ri = (is_response[b * N_ + i] == 1);
  int rj = (is_response[b * N_ + j] == 1);
  int R  = 4 * gt + 2 * ri + rj;

  bool valid = (num_ids[b * N_ + i] >= 0) && (num_ids[b * N_ + j] >= 0) && (i != j);
  float w = valid ? alpha[b * N_ + j] * (1.0f / 127.0f) : 0.0f;

#pragma unroll
  for (int r = 0; r < NREL_; ++r) {
    A16[(((size_t)b * NREL_ + r) * N_ + i) * N_ + j] =
        (_Float16)((r == R) ? w : 0.0f);
  }
}

// ---------------------------------------------------------------------------
// Kernel 2b: pack Init into B-fragment layout for stage-2 GEMM.
// Ipack[(((b*64+ht)*4+kb)*32+lane)*16 + t] = Init16[b][ j=kb*32+8*hi+(t&7)+16*(t>>3) ][ h=ht*16+ln ]
// ---------------------------------------------------------------------------
__global__ void pack_Init_kernel(const _Float16* __restrict__ Init16,
                                 _Float16* __restrict__ Ipack) {
  int gid = blockIdx.x * 256 + threadIdx.x;   // 16*64*4*32*16 = 2^21
  int t    = gid & 15;
  int lane = (gid >> 4) & 31;
  int kb   = (gid >> 9) & 3;
  int ht   = (gid >> 11) & 63;
  int b    = gid >> 17;
  int hi = lane >> 4, ln = lane & 15;
  int j = kb * 32 + hi * 8 + (t & 7) + 16 * (t >> 3);
  int h = ht * 16 + ln;
  Ipack[gid] = Init16[((size_t)b * N_ + j) * H_ + h];
}

// ---------------------------------------------------------------------------
// Kernel 2c: pack W_r (transposed: B[k][n] = W_r[src,n,k]) and w_f
// (B[k][n] = w_f[k,n]) into B-fragment layout for stage-3 GEMM. f32 -> f16.
// Wpack[(((src*64+nt)*32+kb)*32+lane)*16 + t]
// ---------------------------------------------------------------------------
__global__ void pack_W_kernel(const float* __restrict__ W_r,
                              const float* __restrict__ w_f,
                              _Float16* __restrict__ Wpack) {
  int gid = blockIdx.x * 256 + threadIdx.x;   // 9*64*32*32*16 = 9*2^20
  int t    = gid & 15;
  int lane = (gid >> 4) & 31;
  int kb   = (gid >> 9) & 31;
  int nt   = (gid >> 14) & 63;
  int src  = gid >> 20;
  int hi = lane >> 4, ln = lane & 15;
  int k = kb * 32 + hi * 8 + (t & 7) + 16 * (t >> 3);
  int n = nt * 16 + ln;
  float x = (src < NREL_) ? W_r[((size_t)src * H_ + n) * H_ + k]
                          : w_f[(size_t)k * H_ + n];
  Wpack[gid] = (_Float16)x;
}

// ---------------------------------------------------------------------------
// Kernel 3: M16[b,r] (128x1024,f16) = A16[b,r] (128x128) @ Init[b] (128x1024)
// one 16x16 tile per wave, K=128 -> 4 x wmma_f32_16x16x32_f16
// ---------------------------------------------------------------------------
__global__ void gemm_M_kernel(const _Float16* __restrict__ A16,
                              const _Float16* __restrict__ Ipack,
                              _Float16* __restrict__ M16) {
  int tid  = threadIdx.x;          // 128 = 4 waves
  int lane = tid & 31, wave = tid >> 5;
  int ln = lane & 15, hi = lane >> 4;

  int ht = blockIdx.x * 4 + wave;  // 0..63 (16 cols each)
  int it = blockIdx.y;             // 0..7
  int br = blockIdx.z;             // b*8+r
  int b  = br >> 3;

  const _Float16* Arow = A16 + ((size_t)br * N_ + it * 16 + ln) * N_;
  const _Float16* Bbas = Ipack + (((size_t)(b * 64 + ht) * 4) * 32 + lane) * 16;

  v8f acc = {0, 0, 0, 0, 0, 0, 0, 0};
#pragma unroll
  for (int kb = 0; kb < 4; ++kb) {
    v8h a0 = *(const v8h*)(Arow + kb * 32 + 8 * hi);
    v8h a1 = *(const v8h*)(Arow + kb * 32 + 16 + 8 * hi);
    v16h bf = *(const v16h*)(Bbas + (size_t)kb * 512);
    acc = wmma16(cat8(a0, a1), bf, acc);
  }

  _Float16* Mout = M16 + (size_t)br * N_ * H_;
#pragma unroll
  for (int p = 0; p < 8; ++p) {
    int row = it * 16 + p + 8 * hi;
    Mout[(size_t)row * H_ + ht * 16 + ln] = (_Float16)acc[p];
  }
}

// ---------------------------------------------------------------------------
// Kernel 4: out[b] = relu( sum_r M16[b,r] @ W_r^T + Init @ w_f + b_f )
// stacked K = 9*1024; each wave: 2 i-tiles x 4 n-tiles = 8 accumulators.
// Per K-step: 2 A-fragment loads + 4 B-fragment loads -> 8 WMMAs
// (each B fragment feeds 2 WMMAs, each A fragment feeds 4).
// ---------------------------------------------------------------------------
__global__ void gemm_out_kernel(const _Float16* __restrict__ M16,
                                const _Float16* __restrict__ Init16,
                                const _Float16* __restrict__ Wpack,
                                const float* __restrict__ b_f,
                                float* __restrict__ out) {
  int tid  = threadIdx.x;          // 128 = 4 waves
  int lane = tid & 31, wave = tid >> 5;
  int ln = lane & 15, hi = lane >> 4;

  int nt0 = blockIdx.x * 16 + wave * 4;  // 4 n-tiles per wave
  int it0 = blockIdx.y * 2;              // 2 i-tiles per wave (0..7 total)
  int b   = blockIdx.z;                  // 0..15

  v8f acc[2][4];
#pragma unroll
  for (int ii = 0; ii < 2; ++ii)
#pragma unroll
    for (int q = 0; q < 4; ++q) acc[ii][q] = (v8f){0, 0, 0, 0, 0, 0, 0, 0};

  for (int src = 0; src < NREL_ + 1; ++src) {
    const _Float16* Abase =
        (src < NREL_)
            ? M16 + ((size_t)(b * NREL_ + src) * N_ + it0 * 16 + ln) * H_
            : Init16 + ((size_t)b * N_ + it0 * 16 + ln) * H_;
    const _Float16* Wsrc = Wpack + (size_t)src * 64 * 32 * 512;

    for (int kb = 0; kb < 32; ++kb) {
      v16h a[2];
#pragma unroll
      for (int ii = 0; ii < 2; ++ii) {
        const _Float16* Arow = Abase + (size_t)ii * 16 * H_;
        v8h a0 = *(const v8h*)(Arow + kb * 32 + 8 * hi);
        v8h a1 = *(const v8h*)(Arow + kb * 32 + 16 + 8 * hi);
        a[ii] = cat8(a0, a1);
      }
#pragma unroll
      for (int q = 0; q < 4; ++q) {
        const _Float16* bp =
            Wsrc + (((size_t)(nt0 + q) * 32 + kb) * 32 + lane) * 16;
        v16h bf = *(const v16h*)bp;
        acc[0][q] = wmma16(a[0], bf, acc[0][q]);
        acc[1][q] = wmma16(a[1], bf, acc[1][q]);
      }
    }
  }

#pragma unroll
  for (int ii = 0; ii < 2; ++ii) {
#pragma unroll
    for (int q = 0; q < 4; ++q) {
      int col = (nt0 + q) * 16 + ln;
      float bias = b_f[col];
#pragma unroll
      for (int p = 0; p < 8; ++p) {
        int row = (it0 + ii) * 16 + p + 8 * hi;
        out[((size_t)b * N_ + row) * H_ + col] = fmaxf(acc[ii][q][p] + bias, 0.0f);
      }
    }
  }
}

// ---------------------------------------------------------------------------
extern "C" void kernel_launch(void* const* d_in, const int* in_sizes, int n_in,
                              void* d_out, int out_size, void* d_ws,
                              size_t ws_size, hipStream_t stream) {
  (void)in_sizes; (void)n_in; (void)out_size; (void)ws_size;

  const float* word_emb = (const float*)d_in[0];
  const int*   num_ids  = (const int*)d_in[1];
  const int*   is_resp  = (const int*)d_in[2];
  const float* numbers  = (const float*)d_in[3];
  const float* w_alpha  = (const float*)d_in[4];
  const float* b_alpha  = (const float*)d_in[5];
  const float* w_f      = (const float*)d_in[6];
  const float* b_f      = (const float*)d_in[7];
  const float* W_r      = (const float*)d_in[8];
  float* out = (float*)d_out;

  char* ws = (char*)d_ws;
  size_t off = 0;
  _Float16* Init16 = (_Float16*)(ws + off); off += (size_t)B_ * N_ * H_ * 2;          // 4 MB
  float*    alpha  = (float*)(ws + off);    off += (size_t)B_ * N_ * 4;               // 8 KB
  _Float16* A16    = (_Float16*)(ws + off); off += (size_t)B_ * NREL_ * N_ * N_ * 2;  // 4 MB
  _Float16* Ipack  = (_Float16*)(ws + off); off += (size_t)B_ * N_ * H_ * 2;          // 4 MB
  _Float16* Wpack  = (_Float16*)(ws + off); off += (size_t)(NREL_ + 1) * H_ * H_ * 2; // 18.9 MB
  _Float16* M16    = (_Float16*)(ws + off);                                           // 33.5 MB

  gather_alpha_kernel<<<dim3(B_ * N_), dim3(256), 0, stream>>>(
      word_emb, num_ids, w_alpha, b_alpha, Init16, alpha);

  build_A_kernel<<<dim3((B_ * N_ * N_) / 256), dim3(256), 0, stream>>>(
      num_ids, is_resp, numbers, alpha, A16);

  pack_Init_kernel<<<dim3((B_ * N_ * H_) / 256), dim3(256), 0, stream>>>(
      Init16, Ipack);

  pack_W_kernel<<<dim3(((NREL_ + 1) * H_ * H_) / 256), dim3(256), 0, stream>>>(
      W_r, w_f, Wpack);

  gemm_M_kernel<<<dim3(16, 8, B_ * NREL_), dim3(128), 0, stream>>>(
      A16, Ipack, M16);

  gemm_out_kernel<<<dim3(4, 4, B_), dim3(128), 0, stream>>>(
      M16, Init16, Wpack, b_f, out);
}
